// CustomGRU_89593017795423
// MI455X (gfx1250) — 8-bit matrix operand order repaired and checked by reading, compiled, not run on hardware
//
#include <hip/hip_runtime.h>
#include <stdint.h>

#define T_STEPS 128
#define BATCH   64
#define IN_DIM  512
#define HID     1024
#define G3      (3*HID)
#define BH      (BATCH*HID)

typedef int v8i __attribute__((ext_vector_type(8)));

enum {
  SL_X = 0, SL_W, SL_R, SL_BX, SL_BR,
  SL_H    = 8,
  SL_WX   = SL_H    + T_STEPS + 1,
  SL_RH   = SL_WX   + T_STEPS,
  SL_RPRE = SL_RH   + T_STEPS,
  SL_ZPRE = SL_RPRE + T_STEPS,
  SL_RHN  = SL_ZPRE + T_STEPS,
  SL_SR   = SL_RHN  + T_STEPS,
  SL_SZ   = SL_SR   + T_STEPS,
  SL_RRH  = SL_SZ   + T_STEPS,
  SL_GPRE = SL_RRH  + T_STEPS,
  SL_G    = SL_GPRE + T_STEPS,
  SL_OLD  = SL_G    + T_STEPS,
  SL_NEW  = SL_OLD  + T_STEPS,
  SL_COUNT = SL_NEW + T_STEPS
};

__device__ __forceinline__ float slot_scale_p(const unsigned int* p) {
  return fmaxf(__uint_as_float(*p) * (1.0f/127.0f), 1e-12f);
}
__device__ __forceinline__ float fqv(float v, float s) {
  float q = v / s;
  q = fminf(fmaxf(q, -128.0f), 127.0f);
  return rintf(q) * s;
}
__device__ __forceinline__ signed char q8(float v, float s) {
  float q = fminf(fmaxf(v / s, -128.0f), 127.0f);
  return (signed char)(int)rintf(q);
}
__device__ __forceinline__ void wave_amax(float v, unsigned int* slot) {
  #pragma unroll
  for (int off = 16; off > 0; off >>= 1)
    v = fmaxf(v, __shfl_xor(v, off, 32));
  if ((threadIdx.x & 31) == 0) atomicMax(slot, __float_as_uint(v));
}

__global__ void zero_u32_kernel(unsigned int* p, int n) {
  int i = blockIdx.x * blockDim.x + threadIdx.x;
  if (i < n) p[i] = 0u;
}

__global__ void maxabs_kernel(const float* __restrict__ x, long long n,
                              unsigned int* __restrict__ slot) {
  long long stride = (long long)gridDim.x * blockDim.x;
  float m = 0.0f;
  for (long long i = (long long)blockIdx.x * blockDim.x + threadIdx.x; i < n; i += stride)
    m = fmaxf(m, fabsf(x[i]));
  wave_amax(m, slot);
}

__global__ void quant_i8_kernel(const float* __restrict__ x, signed char* __restrict__ q,
                                long long n, const unsigned int* __restrict__ slot) {
  const float s = slot_scale_p(slot);
  long long stride = (long long)gridDim.x * blockDim.x;
  for (long long i = (long long)blockIdx.x * blockDim.x + threadIdx.x; i < n; i += stride)
    q[i] = q8(x[i], s);
}

__global__ void fq_vec_kernel(const float* __restrict__ x, float* __restrict__ o,
                              long long n, const unsigned int* __restrict__ slot) {
  const float s = slot_scale_p(slot);
  long long stride = (long long)gridDim.x * blockDim.x;
  for (long long i = (long long)blockIdx.x * blockDim.x + threadIdx.x; i < n; i += stride)
    o[i] = fqv(x[i], s);
}

__global__ void gemm_i8_wmma_kernel(const signed char* __restrict__ A,
                                    const signed char* __restrict__ Bm,
                                    float* __restrict__ C,
                                    int M, int N, int K,
                                    const unsigned int* __restrict__ sa_slot,
                                    const unsigned int* __restrict__ sb_slot,
                                    unsigned int* __restrict__ omax_slot) {
  const int wave   = (blockIdx.x * blockDim.x + threadIdx.x) >> 5;
  const int lane   = threadIdx.x & 31;
  const int tilesN = N >> 4;
  const int tiles  = (M >> 4) * tilesN;
  if (wave >= tiles) return;
  const int tm = wave / tilesN, tn = wave % tilesN;
  const int m0 = tm << 4, n0 = tn << 4;
  const int hi = lane >> 4;

  const signed char* arow = A  + (long long)(m0 + (lane & 15)) * K + (hi ? 16 : 0);
  const signed char* brow = Bm + (long long)(n0 + (lane & 15)) * K + (hi ? 16 : 0);

  v8i acc = {0, 0, 0, 0, 0, 0, 0, 0};
  for (int k0 = 0; k0 < K; k0 += 64) {
    __builtin_prefetch(arow + k0 + 256, 0, 0);
    __builtin_prefetch(brow + k0 + 256, 0, 0);
    v8i a, b;
    const int2 a0 = *(const int2*)(arow + k0);
    const int2 a1 = *(const int2*)(arow + k0 + 8);
    const int2 a2 = *(const int2*)(arow + k0 + 32);
    const int2 a3 = *(const int2*)(arow + k0 + 40);
    a[0] = a0.x; a[1] = a0.y; a[2] = a1.x; a[3] = a1.y;
    a[4] = a2.x; a[5] = a2.y; a[6] = a3.x; a[7] = a3.y;
    const int4 b0 = *(const int4*)(brow + k0);
    const int4 b1 = *(const int4*)(brow + k0 + 32);
    b[0] = b0.x; b[1] = b0.y; b[2] = b0.z; b[3] = b0.w;
    b[4] = b1.x; b[5] = b1.y; b[6] = b1.z; b[7] = b1.w;
    acc = __builtin_amdgcn_wmma_i32_16x16x64_iu8(true, a, true, b, acc, false, false);
  }

  const float s = slot_scale_p(sa_slot) * slot_scale_p(sb_slot);
  float lmax = 0.0f;
  float* crow = C + (long long)(m0 + hi * 8) * N + n0 + (lane & 15);
  #pragma unroll
  for (int i = 0; i < 8; ++i) {
    float v = (float)acc[i] * s;
    crow[(long long)i * N] = v;
    lmax = fmaxf(lmax, fabsf(v));
  }
  wave_amax(lmax, omax_slot);
}

__global__ void gru_gates1_kernel(const float* __restrict__ wxraw, const float* __restrict__ rhraw,
                                  const float* __restrict__ qbx, const float* __restrict__ qbr,
                                  float* __restrict__ rpre, float* __restrict__ zpre,
                                  float* __restrict__ rhn, unsigned int* scl, int t) {
  const float swx = slot_scale_p(scl + SL_WX + t);
  const float srh = slot_scale_p(scl + SL_RH + t);
  float mr = 0.f, mz = 0.f, mn = 0.f;
  for (int idx = blockIdx.x * blockDim.x + threadIdx.x; idx < BH; idx += gridDim.x * blockDim.x) {
    int b = idx >> 10, j = idx & (HID - 1);
    long long base = (long long)b * G3 + j;
    float rp = fqv(wxraw[base],         swx) + qbx[j]         + fqv(rhraw[base],         srh) + qbr[j];
    float zp = fqv(wxraw[base + HID],   swx) + qbx[HID + j]   + fqv(rhraw[base + HID],   srh) + qbr[HID + j];
    float rn = fqv(rhraw[base + 2*HID], srh) + qbr[2*HID + j];
    rpre[idx] = rp; zpre[idx] = zp; rhn[idx] = rn;
    mr = fmaxf(mr, fabsf(rp)); mz = fmaxf(mz, fabsf(zp)); mn = fmaxf(mn, fabsf(rn));
  }
  wave_amax(mr, scl + SL_RPRE + t);
  wave_amax(mz, scl + SL_ZPRE + t);
  wave_amax(mn, scl + SL_RHN  + t);
}

__global__ void gru_sigmoid_kernel(const float* __restrict__ rpre, const float* __restrict__ zpre,
                                   float* __restrict__ srb, float* __restrict__ szb,
                                   unsigned int* scl, int t) {
  const float sp  = slot_scale_p(scl + SL_RPRE + t);
  const float szp = slot_scale_p(scl + SL_ZPRE + t);
  float mr = 0.f, mz = 0.f;
  for (int idx = blockIdx.x * blockDim.x + threadIdx.x; idx < BH; idx += gridDim.x * blockDim.x) {
    float r = 1.0f / (1.0f + expf(-fqv(rpre[idx], sp)));
    float z = 1.0f / (1.0f + expf(-fqv(zpre[idx], szp)));
    srb[idx] = r; szb[idx] = z;
    mr = fmaxf(mr, fabsf(r)); mz = fmaxf(mz, fabsf(z));
  }
  wave_amax(mr, scl + SL_SR + t);
  wave_amax(mz, scl + SL_SZ + t);
}

__global__ void gru_rrh_kernel(const float* __restrict__ srb, const float* __restrict__ rhn,
                               float* __restrict__ rrh, unsigned int* scl, int t) {
  const float ssr  = slot_scale_p(scl + SL_SR  + t);
  const float srhn = slot_scale_p(scl + SL_RHN + t);
  float m = 0.f;
  for (int idx = blockIdx.x * blockDim.x + threadIdx.x; idx < BH; idx += gridDim.x * blockDim.x) {
    float v = fqv(srb[idx], ssr) * fqv(rhn[idx], srhn);
    rrh[idx] = v; m = fmaxf(m, fabsf(v));
  }
  wave_amax(m, scl + SL_RRH + t);
}

__global__ void gru_gpre_kernel(const float* __restrict__ wxraw, const float* __restrict__ qbx,
                                const float* __restrict__ rrh, float* __restrict__ gpre,
                                unsigned int* scl, int t) {
  const float swx  = slot_scale_p(scl + SL_WX  + t);
  const float srrh = slot_scale_p(scl + SL_RRH + t);
  float m = 0.f;
  for (int idx = blockIdx.x * blockDim.x + threadIdx.x; idx < BH; idx += gridDim.x * blockDim.x) {
    int b = idx >> 10, j = idx & (HID - 1);
    float v = fqv(wxraw[(long long)b * G3 + 2*HID + j], swx) + qbx[2*HID + j] + fqv(rrh[idx], srrh);
    gpre[idx] = v; m = fmaxf(m, fabsf(v));
  }
  wave_amax(m, scl + SL_GPRE + t);
}

__global__ void gru_tanh_old_kernel(const float* __restrict__ gpre, const float* __restrict__ szb,
                                    const float* __restrict__ h, float* __restrict__ gt,
                                    float* __restrict__ oldc, unsigned int* scl, int t) {
  const float sg = slot_scale_p(scl + SL_GPRE + t);
  const float sz = slot_scale_p(scl + SL_SZ   + t);
  float mg = 0.f, mo = 0.f;
  for (int idx = blockIdx.x * blockDim.x + threadIdx.x; idx < BH; idx += gridDim.x * blockDim.x) {
    float g = tanhf(fqv(gpre[idx], sg));
    float o = fqv(szb[idx], sz) * h[idx];
    gt[idx] = g; oldc[idx] = o;
    mg = fmaxf(mg, fabsf(g)); mo = fmaxf(mo, fabsf(o));
  }
  wave_amax(mg, scl + SL_G   + t);
  wave_amax(mo, scl + SL_OLD + t);
}

__global__ void gru_new_kernel(const float* __restrict__ szb, const float* __restrict__ gt,
                               float* __restrict__ newc, unsigned int* scl, int t) {
  const float sz = slot_scale_p(scl + SL_SZ + t);
  const float sg = slot_scale_p(scl + SL_G  + t);
  float m = 0.f;
  for (int idx = blockIdx.x * blockDim.x + threadIdx.x; idx < BH; idx += gridDim.x * blockDim.x) {
    float v = (1.0f - fqv(szb[idx], sz)) * fqv(gt[idx], sg);
    newc[idx] = v; m = fmaxf(m, fabsf(v));
  }
  wave_amax(m, scl + SL_NEW + t);
}

__global__ void gru_update_kernel(const float* __restrict__ oldc, const float* __restrict__ newc,
                                  float* __restrict__ h, float* __restrict__ out,
                                  unsigned int* scl, int t) {
  const float so = slot_scale_p(scl + SL_OLD + t);
  const float sn = slot_scale_p(scl + SL_NEW + t);
  float m = 0.f;
  for (int idx = blockIdx.x * blockDim.x + threadIdx.x; idx < BH; idx += gridDim.x * blockDim.x) {
    float hn = fqv(oldc[idx], so) + fqv(newc[idx], sn);
    h[idx] = hn;
    out[(long long)t * BH + idx] = hn;
    if (t == T_STEPS - 1) out[(long long)T_STEPS * BH + idx] = hn;
    m = fmaxf(m, fabsf(hn));
  }
  wave_amax(m, scl + SL_H + t + 1);
}

extern "C" void kernel_launch(void* const* d_in, const int* in_sizes, int n_in,
                              void* d_out, int out_size, void* d_ws, size_t ws_size,
                              hipStream_t stream) {
  (void)in_sizes; (void)n_in; (void)out_size; (void)ws_size;
  const float* x  = (const float*)d_in[0];
  const float* h0 = (const float*)d_in[1];
  const float* W  = (const float*)d_in[2];
  const float* R  = (const float*)d_in[3];
  const float* bx = (const float*)d_in[4];
  const float* br = (const float*)d_in[5];
  float* out = (float*)d_out;

  char* ws = (char*)d_ws;
  size_t off = 0;
  auto alloc = [&](size_t bytes) -> char* {
    char* p = ws + off; off += (bytes + 255) & ~(size_t)255; return p;
  };
  unsigned int* scl = (unsigned int*)alloc(SL_COUNT * sizeof(unsigned int));
  signed char*  Wq  = (signed char*)alloc((size_t)G3 * IN_DIM);
  signed char*  Rq  = (signed char*)alloc((size_t)G3 * HID);
  signed char*  xq  = (signed char*)alloc((size_t)T_STEPS * BATCH * IN_DIM);
  float* qbx  = (float*)alloc((size_t)G3 * 4);
  float* qbr  = (float*)alloc((size_t)G3 * 4);
  float* h    = (float*)alloc((size_t)BH * 4);
  signed char* hq = (signed char*)alloc((size_t)BH);
  float* wxraw = (float*)alloc((size_t)BATCH * G3 * 4);
  float* rhraw = (float*)alloc((size_t)BATCH * G3 * 4);
  float* rpre = (float*)alloc((size_t)BH * 4);
  float* zpre = (float*)alloc((size_t)BH * 4);
  float* rhn  = (float*)alloc((size_t)BH * 4);
  float* srb  = (float*)alloc((size_t)BH * 4);
  float* szb  = (float*)alloc((size_t)BH * 4);
  float* rrh  = (float*)alloc((size_t)BH * 4);
  float* gpre = (float*)alloc((size_t)BH * 4);
  float* gt   = (float*)alloc((size_t)BH * 4);
  float* oldc = (float*)alloc((size_t)BH * 4);
  float* newc = (float*)alloc((size_t)BH * 4);

  zero_u32_kernel<<<(SL_COUNT + 255) / 256, 256, 0, stream>>>(scl, SL_COUNT);
  maxabs_kernel<<<512, 256, 0, stream>>>(x,  (long long)T_STEPS * BATCH * IN_DIM, scl + SL_X);
  maxabs_kernel<<<512, 256, 0, stream>>>(W,  (long long)G3 * IN_DIM,              scl + SL_W);
  maxabs_kernel<<<512, 256, 0, stream>>>(R,  (long long)G3 * HID,                 scl + SL_R);
  maxabs_kernel<<<12,  256, 0, stream>>>(bx, (long long)G3,                       scl + SL_BX);
  maxabs_kernel<<<12,  256, 0, stream>>>(br, (long long)G3,                       scl + SL_BR);
  maxabs_kernel<<<256, 256, 0, stream>>>(h0, (long long)BH,                       scl + SL_H);
  quant_i8_kernel<<<512, 256, 0, stream>>>(x, xq, (long long)T_STEPS * BATCH * IN_DIM, scl + SL_X);
  quant_i8_kernel<<<512, 256, 0, stream>>>(W, Wq, (long long)G3 * IN_DIM,              scl + SL_W);
  quant_i8_kernel<<<512, 256, 0, stream>>>(R, Rq, (long long)G3 * HID,                 scl + SL_R);
  fq_vec_kernel<<<12, 256, 0, stream>>>(bx, qbx, (long long)G3, scl + SL_BX);
  fq_vec_kernel<<<12, 256, 0, stream>>>(br, qbr, (long long)G3, scl + SL_BR);
  hipMemcpyAsync(h, h0, (size_t)BH * 4, hipMemcpyDeviceToDevice, stream);

  const int tiles = (BATCH / 16) * (G3 / 16);
  const int gemm_blocks = tiles * 32 / 256;
  const dim3 eb(256), eg(BH / 256);

  for (int t = 0; t < T_STEPS; ++t) {
    gemm_i8_wmma_kernel<<<gemm_blocks, 256, 0, stream>>>(
        xq + (size_t)t * BATCH * IN_DIM, Wq, wxraw, BATCH, G3, IN_DIM,
        scl + SL_X, scl + SL_W, scl + SL_WX + t);
    quant_i8_kernel<<<64, 256, 0, stream>>>(h, hq, (long long)BH, scl + SL_H + t);
    gemm_i8_wmma_kernel<<<gemm_blocks, 256, 0, stream>>>(
        hq, Rq, rhraw, BATCH, G3, HID,
        scl + SL_H + t, scl + SL_R, scl + SL_RH + t);
    gru_gates1_kernel <<<eg, eb, 0, stream>>>(wxraw, rhraw, qbx, qbr, rpre, zpre, rhn, scl, t);
    gru_sigmoid_kernel<<<eg, eb, 0, stream>>>(rpre, zpre, srb, szb, scl, t);
    gru_rrh_kernel    <<<eg, eb, 0, stream>>>(srb, rhn, rrh, scl, t);
    gru_gpre_kernel   <<<eg, eb, 0, stream>>>(wxraw, qbx, rrh, gpre, scl, t);
    gru_tanh_old_kernel<<<eg, eb, 0, stream>>>(gpre, szb, h, gt, oldc, scl, t);
    gru_new_kernel    <<<eg, eb, 0, stream>>>(szb, gt, newc, scl, t);
    gru_update_kernel <<<eg, eb, 0, stream>>>(oldc, newc, h, out, scl, t);
  }
}
